// SDF_17059610099800
// MI455X (gfx1250) — matrix operand order repaired and checked by reading, not compiled, not run on hardware
//
#include <hip/hip_runtime.h>

typedef __attribute__((ext_vector_type(16))) _Float16 v16h;
typedef __attribute__((ext_vector_type(8)))  float    v8f;

#define NRAYS 8192
#define TITER 8
#define SDF_THR 5e-5f
#define FIN_THR 0.0015625f   // (BMAX-BMIN)/10/RES = 2/10/128

#define WAVES_PER_BLOCK 2
#define PTS_PER_WAVE    32           // two 16-row WMMA tiles per wave
#define PTS_PER_BLOCK   (WAVES_PER_BLOCK * PTS_PER_WAVE)
#define FRAG_BYTES      1024
#define HID_OFF         0            // 8 K-tiles * 1024B
#define ENC_OFF         8192         // 2 K-tiles * 1024B
#define SET_LDS         10240        // one 16-row activation set
#define WAVE_LDS        (2 * SET_LDS)

// ---------------- math helpers ----------------
// softplus(100x)/100, branchless, raw HW exp2/log2 (1+e^-|z| is in [1,2]):
// log1p(e^z) = max(z,0) + log1p(e^-|z|)
__device__ __forceinline__ float softplus100(float x) {
    float z = 100.0f * x;
    float e = __builtin_amdgcn_exp2f(-fabsf(z) * 1.44269504f);       // e^{-|z|}
    float l = __builtin_amdgcn_logf(1.0f + e) * 0.6931471806f;       // log1p(e^{-|z|})
    return 0.01f * (fmaxf(z, 0.0f) + l);
}

// A-matrix 16x32 f16 fragment mapping (ISA 7.12.2): element (m, klocal) ->
// lane' = m + 16*g, half-index e.  Branchless:
//   hi = klocal>>4, low = klocal&15, g = (low>>3)&1, e = 8*hi + (low&7)
__device__ __forceinline__ void amap(int klocal, int& g, int& e) {
    int hi = klocal >> 4, low = klocal & 15;
    g = (low >> 3) & 1;
    e = 8 * hi + (low & 7);
}

// ---------------- weight packing ----------------
// B-matrix 32x16 f16 fragment: lane' = n + 16*((klocal>>3)&1), e = 8*(klocal>>4) + (klocal&7).
// Packed per layer: fragment index (nt*KT + kt), 512 halves each, lane-major
// (lane'*16 + e) so eval loads 32B contiguous per lane (2x global_load_b128).
__global__ void k_pack(const float* __restrict__ w, _Float16* __restrict__ dst,
                       int din, int KT) {
    int idx = blockIdx.x * 256 + threadIdx.x;
    if (idx >= KT * 8192) return;          // KT*32 rows x 256 cols
    int n  = idx & 255;
    int k  = idx >> 8;
    int kt = k >> 5, kl = k & 31;
    int nt = n >> 4, nl = n & 15;
    size_t off = ((size_t)(nt * KT + kt) << 9)
               + ((size_t)(nl + (((kl >> 3) & 1) << 4)) << 4) + (((kl >> 4) << 3) + (kl & 7));
    dst[off] = (_Float16)((k < din) ? w[(size_t)k * 256 + n] : 0.0f);
}

// w8[:,0] packed in A-fragment lane layout (matches how h7 sits in LDS):
// dst[kt*512 + lane*16 + e] = w8[(kt*32 + klocal(lane,e)) * 257]
__global__ void k_packw8(const float* __restrict__ w8, _Float16* __restrict__ dst) {
    int idx = blockIdx.x * 256 + threadIdx.x;
    if (idx >= 4096) return;
    int kt = idx >> 9, r = idx & 511, lane = r >> 4, e = r & 15;
    int g = lane >> 4;
    int kl = (e < 8) ? (8 * g + e) : (16 + 8 * g + (e - 8));
    dst[idx] = (_Float16)w8[(size_t)(kt * 32 + kl) * 257];
}

// ---------------- MLP layer (WMMA core) ----------------
// One wave drives TWO 16-row tiles: each B fragment is loaded once and feeds
// two WMMAs (halves weight-stream traffic, doubles FLOP/byte from cache).
template <int KT, int ENCT>
__device__ __forceinline__ void run_layer(unsigned char* ldsW, int aOff,
                                          const _Float16* __restrict__ wl,
                                          const float* __restrict__ bias, int lane) {
    v16h a0[KT], a1[KT];
#pragma unroll
    for (int kt = 0; kt < KT; ++kt) {
        int off = (kt < KT - ENCT) ? (aOff + kt * FRAG_BYTES)
                                   : (ENC_OFF + (kt - (KT - ENCT)) * FRAG_BYTES);
        a0[kt] = *(const v16h*)(ldsW + off + lane * 32);
        a1[kt] = *(const v16h*)(ldsW + SET_LDS + off + lane * 32);
    }
    const int n   = lane & 15;
    const int hi8 = (lane >> 4) * 8;
    const int g   = (n >> 3) & 1;                 // store-map g: depends only on lane
    unsigned char* dstLane0 = ldsW + HID_OFF + (16 * g + hi8) * 32 + (n & 7) * 2;
    unsigned char* dstLane1 = dstLane0 + SET_LDS;
#pragma unroll 1                                   // keep I$-resident
    for (int nt = 0; nt < 16; ++nt) {
        v8f acc0 = {0.f, 0.f, 0.f, 0.f, 0.f, 0.f, 0.f, 0.f};
        v8f acc1 = {0.f, 0.f, 0.f, 0.f, 0.f, 0.f, 0.f, 0.f};
        const _Float16* wf = wl + (size_t)nt * KT * 512 + lane * 16;
#pragma unroll
        for (int kt = 0; kt < KT; ++kt) {
            v16h b = *(const v16h*)(wf + kt * 512);
            acc0 = __builtin_amdgcn_wmma_f32_16x16x32_f16(
                false, a0[kt], false, b, (short)0, acc0, false, false);
            acc1 = __builtin_amdgcn_wmma_f32_16x16x32_f16(
                false, a1[kt], false, b, (short)0, acc1, false, false);
        }
        float bv = bias[nt * 16 + n];
        int ntOff = (nt >> 1) * FRAG_BYTES + (nt & 1) * 16;
        unsigned char* dst0 = dstLane0 + ntOff;
        unsigned char* dst1 = dstLane1 + ntOff;
#pragma unroll
        for (int r = 0; r < 8; ++r) {
            *(_Float16*)(dst0 + r * 32) = (_Float16)softplus100(acc0[r] + bv);
            *(_Float16*)(dst1 + r * 32) = (_Float16)softplus100(acc1[r] + bv);
        }
    }
}

// ---------------- full MLP eval: pts (M,3) -> sdf (M,) ----------------
struct EvalArgs {
    const float* pts;
    float*       out;
    const _Float16* wp;       // packed weights, layers 0..7
    const _Float16* w8p;      // packed w8[:,0], fragment lane layout (4096 halves)
    const float* b8;          // b8 vector (use [0])
    const float* bias[8];
};

__launch_bounds__(WAVES_PER_BLOCK * 32)
__global__ void k_eval(EvalArgs A) {
    __shared__ __align__(32) unsigned char smem[WAVES_PER_BLOCK * WAVE_LDS];
    const int tid  = threadIdx.x;
    const int wave = tid >> 5;
    const int lane = tid & 31;
    unsigned char* ldsW = smem + wave * WAVE_LDS;
    const int ptBase = blockIdx.x * PTS_PER_BLOCK + wave * PTS_PER_WAVE;

    // ---- positional encoding, stored directly in A-fragment layout ----
    // all 32 lanes: lanes 0-15 -> set0 rows, lanes 16-31 -> set1 rows
    {
        const float* p = A.pts + (size_t)(ptBase + lane) * 3;
        unsigned char* setB = ldsW + (lane >> 4) * SET_LDS;
        const int m = lane & 15;
        float x = p[0], y = p[1], z = p[2];  // xn == xyz for BMIN=-1,BMAX=1,RESCALE=1
        float enc[64];
        enc[0] = x; enc[1] = y; enc[2] = z;
        float f = 1.0f;
#pragma unroll
        for (int l = 0; l < 6; ++l) {
            enc[3 + l * 3 + 0]  = __sinf(x * f);
            enc[3 + l * 3 + 1]  = __sinf(y * f);
            enc[3 + l * 3 + 2]  = __sinf(z * f);
            enc[21 + l * 3 + 0] = __cosf(x * f);
            enc[21 + l * 3 + 1] = __cosf(y * f);
            enc[21 + l * 3 + 2] = __cosf(z * f);
            f *= 2.0f;
        }
#pragma unroll
        for (int k = 39; k < 64; ++k) enc[k] = 0.0f;
#pragma unroll
        for (int k = 0; k < 64; ++k) {
            int g, e; amap(k & 31, g, e);
            *(_Float16*)(setB + ENC_OFF + (k >> 5) * FRAG_BYTES
                         + (m + 16 * g) * 32 + e * 2) = (_Float16)enc[k];
        }
    }
    // same-wave DS ops are in-order: no barrier needed (per-wave private LDS)

    constexpr size_t W0 = 0, W1 = 16384, W2 = 81920,  W3 = 147456;
    constexpr size_t W4 = 212992, W5 = 294912, W6 = 360448, W7 = 425984;

    run_layer<2, 0>(ldsW, ENC_OFF, A.wp + W0, A.bias[0], lane);
    run_layer<8, 0>(ldsW, HID_OFF, A.wp + W1, A.bias[1], lane);
    run_layer<8, 0>(ldsW, HID_OFF, A.wp + W2, A.bias[2], lane);
    run_layer<8, 0>(ldsW, HID_OFF, A.wp + W3, A.bias[3], lane);
    run_layer<10, 2>(ldsW, HID_OFF, A.wp + W4, A.bias[4], lane);  // skip concat
    run_layer<8, 0>(ldsW, HID_OFF, A.wp + W5, A.bias[5], lane);
    run_layer<8, 0>(ldsW, HID_OFF, A.wp + W6, A.bias[6], lane);
    run_layer<8, 0>(ldsW, HID_OFF, A.wp + W7, A.bias[7], lane);

    // ---- final layer, channel 0 only: dot(h7, w8[:,0]) + b8[0] ----
    float part0 = 0.0f, part1 = 0.0f;
#pragma unroll
    for (int kt = 0; kt < 8; ++kt) {
        v16h h0 = *(const v16h*)(ldsW + HID_OFF + kt * FRAG_BYTES + lane * 32);
        v16h h1 = *(const v16h*)(ldsW + SET_LDS + HID_OFF + kt * FRAG_BYTES + lane * 32);
        v16h wv = *(const v16h*)(A.w8p + kt * 512 + lane * 16);
#pragma unroll
        for (int e = 0; e < 16; ++e) {
            part0 += (float)h0[e] * (float)wv[e];
            part1 += (float)h1[e] * (float)wv[e];
        }
    }
    part0 += __shfl_xor(part0, 16, 32);   // combine the two half-rows
    part1 += __shfl_xor(part1, 16, 32);
    if (lane < 16) {
        float b = A.b8[0];
        A.out[ptBase + lane]      = part0 + b;
        A.out[ptBase + 16 + lane] = part1 + b;
    }
}

// ---------------- sphere-trace drivers ----------------
// state floats: [0]=tmin [N]=tmax [2N]=acc_s [3N]=acc_e [4N]=next_s [5N]=next_e
// mask ints:    [0]=mask_s [N]=mask_e [2N]=pend_s [3N]=pend_e
__global__ void k_init(const float* __restrict__ o3, const float* __restrict__ d3,
                       float* __restrict__ pts, float* __restrict__ st,
                       int* __restrict__ msk) {
    int i = blockIdx.x * blockDim.x + threadIdx.x;
    if (i >= NRAYS) return;
    float o[3], d[3];
#pragma unroll
    for (int c = 0; c < 3; ++c) { o[c] = o3[i * 3 + c]; d[c] = d3[i * 3 + c]; }
    float tmin = -1e30f, tmax = 1e30f;
#pragma unroll
    for (int c = 0; c < 3; ++c) {
        float inv = 1.0f / d[c];
        float t1 = (-1.0f - o[c]) * inv, t2 = (1.0f - o[c]) * inv;
        tmin = fmaxf(tmin, fminf(t1, t2));
        tmax = fminf(tmax, fmaxf(t1, t2));
    }
    tmin = fmaxf(tmin, 0.0f);
#pragma unroll
    for (int c = 0; c < 3; ++c) {
        pts[(size_t)i * 3 + c]           = o[c] + tmin * d[c];
        pts[(size_t)(NRAYS + i) * 3 + c] = o[c] + tmax * d[c];
    }
    st[i] = tmin; st[NRAYS + i] = tmax;
    st[2 * NRAYS + i] = tmin; st[3 * NRAYS + i] = tmax;
    st[4 * NRAYS + i] = 0.0f; st[5 * NRAYS + i] = 0.0f;
    msk[i] = 1; msk[NRAYS + i] = 1; msk[2 * NRAYS + i] = 1; msk[3 * NRAYS + i] = 1;
}

__global__ void k_step(const float* __restrict__ o3, const float* __restrict__ d3,
                       const float* __restrict__ sdfB, float* __restrict__ pts,
                       float* __restrict__ track, float* __restrict__ st,
                       int* __restrict__ msk, int t) {
    int i = blockIdx.x * blockDim.x + threadIdx.x;
    if (i >= NRAYS) return;
    float tmax  = st[NRAYS + i];
    float acc_s = st[2 * NRAYS + i], acc_e = st[3 * NRAYS + i];
    float next_s = st[4 * NRAYS + i], next_e = st[5 * NRAYS + i];
    int ms = msk[i], me = msk[NRAYS + i];
    int ps = msk[2 * NRAYS + i], pe = msk[3 * NRAYS + i];
    // merge pending masked evaluations (reference: next_x = where(mask_x, sdf, next_x))
    next_s = ps ? sdfB[i]          : next_s;
    next_e = pe ? sdfB[NRAYS + i]  : next_e;
    float curr_s = (fabsf(next_s) <= SDF_THR) ? 0.0f : next_s;
    float curr_e = (fabsf(next_e) <= SDF_THR) ? 0.0f : next_e;
    ms = ms && (fabsf(curr_s) > SDF_THR);
    me = me && (fabsf(curr_e) > SDF_THR);
    float o[3], d[3];
#pragma unroll
    for (int c = 0; c < 3; ++c) { o[c] = o3[i * 3 + c]; d[c] = d3[i * 3 + c]; }
#pragma unroll
    for (int c = 0; c < 3; ++c)   // append track point BEFORE the march
        track[(size_t)(i * TITER + t) * 3 + c] = o[c] + acc_s * d[c];
    acc_s = fminf(acc_s + curr_s, tmax);
    acc_e = fminf(acc_e + curr_e, tmax);
#pragma unroll
    for (int c = 0; c < 3; ++c) {
        pts[(size_t)i * 3 + c]           = o[c] + acc_s * d[c];
        pts[(size_t)(NRAYS + i) * 3 + c] = o[c] + acc_e * d[c];
    }
    msk[2 * NRAYS + i] = ms;   // intra-iteration masks gate the NEXT merge
    msk[3 * NRAYS + i] = me;
    int ok = acc_s < acc_e;
    msk[i] = ms && ok; msk[NRAYS + i] = me && ok;
    st[2 * NRAYS + i] = acc_s; st[3 * NRAYS + i] = acc_e;
    st[4 * NRAYS + i] = next_s; st[5 * NRAYS + i] = next_e;
}

// out layout (floats): [0,N) d_pred | [N,2N) last sdf |
// [2N, 2N+4096*24) pts_tracks[pick] | then [8192*3) sampled | [N) finish mask
#define O_SPTS  (2 * NRAYS)
#define O_SAMP  (O_SPTS + 4096 * 24)
#define O_MASK  (O_SAMP + NRAYS * 3)
__global__ void k_final(const float* __restrict__ o3, const float* __restrict__ d3,
                        const float* __restrict__ frand, const int* __restrict__ pick,
                        const float* __restrict__ strack, const float* __restrict__ st,
                        const float* __restrict__ track, float* __restrict__ out) {
    int i = blockIdx.x * blockDim.x + threadIdx.x;
    if (i >= NRAYS) return;
    float tmin = st[i], tmax = st[NRAYS + i], acc_e = st[3 * NRAYS + i];
    float ssum = 0.0f;
#pragma unroll
    for (int t = 0; t < TITER; ++t) ssum += strack[i * TITER + t];
    out[i] = fminf(ssum + tmin, tmax);
    float last = strack[i * TITER + (TITER - 1)];
    out[NRAYS + i] = last;
    float d_up = fminf(1.5f * acc_e, tmax);
    float fr = frand[i];
    float ds = (1.0f - fr) * d_up + fr * tmin;
#pragma unroll
    for (int c = 0; c < 3; ++c)
        out[O_SAMP + i * 3 + c] = o3[i * 3 + c] + ds * d3[i * 3 + c];
    out[O_MASK + i] = (fabsf(last) < FIN_THR) ? 1.0f : 0.0f;
    if (i < 4096) {
        int r = pick[i];
#pragma unroll
        for (int t = 0; t < TITER; ++t)
#pragma unroll
            for (int c = 0; c < 3; ++c)
                out[O_SPTS + i * 24 + t * 3 + c] = track[(size_t)(r * TITER + t) * 3 + c];
    }
}

// ---------------- launcher ----------------
extern "C" void kernel_launch(void* const* d_in, const int* in_sizes, int n_in,
                              void* d_out, int out_size, void* d_ws, size_t ws_size,
                              hipStream_t stream) {
    (void)in_sizes; (void)n_in; (void)out_size; (void)ws_size;
    const float* ray0  = (const float*)d_in[0];
    const float* rdir  = (const float*)d_in[1];
    const float* frand = (const float*)d_in[2];
    const int*   pick  = (const int*)d_in[3];
    const float* W[9]; const float* Bv[9];
    for (int i = 0; i < 9; ++i) { W[i] = (const float*)d_in[4 + 2 * i]; Bv[i] = (const float*)d_in[5 + 2 * i]; }

    char* ws = (char*)d_ws;
    const size_t OFF_W   = 0;         // packed f16 weights: 983040 B
    const size_t OFF_W8  = 983040;    // packed f16 w8 col0: 8192 B
    const size_t OFF_PTS = 991232;    // 2N*3 f32 = 196608 B
    const size_t OFF_SDF = 1187840;   // 2N f32   = 65536 B
    const size_t OFF_TRK = 1253376;   // N*8*3 f32 = 786432 B
    const size_t OFF_STK = 2039808;   // N*8 f32   = 262144 B
    const size_t OFF_ST  = 2301952;   // 6N f32    = 196608 B
    const size_t OFF_MSK = 2498560;   // 4N i32    = 131072 B

    _Float16* wp   = (_Float16*)(ws + OFF_W);
    _Float16* w8p  = (_Float16*)(ws + OFF_W8);
    float* ptsB    = (float*)(ws + OFF_PTS);
    float* sdfB    = (float*)(ws + OFF_SDF);
    float* track   = (float*)(ws + OFF_TRK);
    float* strack  = (float*)(ws + OFF_STK);
    float* st      = (float*)(ws + OFF_ST);
    int*   msk     = (int*)(ws + OFF_MSK);

    const int    KTs[8]   = {2, 8, 8, 8, 10, 8, 8, 8};
    const int    dins[8]  = {39, 256, 256, 256, 295, 256, 256, 256};
    const size_t wOffH[8] = {0, 16384, 81920, 147456, 212992, 294912, 360448, 425984};
    for (int l = 0; l < 8; ++l) {
        int tot = KTs[l] * 8192;
        k_pack<<<(tot + 255) / 256, 256, 0, stream>>>(W[l], wp + wOffH[l], dins[l], KTs[l]);
    }
    k_packw8<<<16, 256, 0, stream>>>(W[8], w8p);

    k_init<<<NRAYS / 256, 256, 0, stream>>>(ray0, rdir, ptsB, st, msk);

    EvalArgs ea;
    ea.wp = wp; ea.w8p = w8p; ea.b8 = Bv[8];
    for (int i = 0; i < 8; ++i) ea.bias[i] = Bv[i];

    ea.pts = ptsB; ea.out = sdfB;
    k_eval<<<(2 * NRAYS) / PTS_PER_BLOCK, WAVES_PER_BLOCK * 32, 0, stream>>>(ea);
    for (int t = 0; t < TITER; ++t) {
        k_step<<<NRAYS / 256, 256, 0, stream>>>(ray0, rdir, sdfB, ptsB, track, st, msk, t);
        if (t < TITER - 1)   // eval after the last step is never consumed
            k_eval<<<(2 * NRAYS) / PTS_PER_BLOCK, WAVES_PER_BLOCK * 32, 0, stream>>>(ea);
    }
    ea.pts = track; ea.out = strack;
    k_eval<<<(NRAYS * TITER) / PTS_PER_BLOCK, WAVES_PER_BLOCK * 32, 0, stream>>>(ea);

    k_final<<<NRAYS / 256, 256, 0, stream>>>(ray0, rdir, frand, pick, strack, st,
                                             track, (float*)d_out);
}
